// EncLayer_14422500180019
// MI455X (gfx1250) — matrix operand order repaired and checked by reading, not compiled, not run on hardware
//
#include <hip/hip_runtime.h>
#include <hip/hip_bf16.h>

typedef __attribute__((ext_vector_type(16))) _Float16 v16h;
typedef __attribute__((ext_vector_type(8)))  float    v8f;
typedef __attribute__((ext_vector_type(4)))  unsigned int u32x4;

#define NNODE 12000
#define KNB   32
#define HD    128
#define FIN   384          // H + CE + H
#define KT_FIN 12          // 384 / 32
#define KT_H   4           // 128 / 32
#define KT_FFN 16          // 512 / 32

// packed-weight offsets in halves inside d_ws
#define OFF_W1   0
#define OFF_W2   49152
#define OFF_W3   65536
#define OFF_W11  81920
#define OFF_W12  131072
#define OFF_W13  147456
#define OFF_WIN  163840
#define OFF_WOUT 229376

__device__ __forceinline__ float gelu_erf(float x) {
    return 0.5f * x * (1.0f + erff(x * 0.70710678118654752f));
}

// D = A(16x32 tiles, from LDS f16) x B(packed fragments in global) summed over KT ktiles
__device__ __forceinline__ v8f mma_rowtile(const _Float16* sA, int lda, int rowBase,
                                           const _Float16* Wp, int KT, int nt, int lane) {
    const int m  = lane & 15;
    const int kb = (lane & 16) ? 8 : 0;   // A layout: lanes 16-31 start at K=8 / K=24
    const _Float16* arow = sA + (rowBase + m) * lda + kb;
    v8f acc;
#pragma unroll
    for (int j = 0; j < 8; ++j) acc[j] = 0.0f;
    for (int kt = 0; kt < KT; ++kt) {
        union { v16h v; u32x4 q[2]; } a;
        a.q[0] = *(const u32x4*)(arow + kt * 32);        // K = kb .. kb+7
        a.q[1] = *(const u32x4*)(arow + kt * 32 + 16);   // K = kb+16 .. kb+23
        v16h b = *(const v16h*)(Wp + ((size_t)(nt * KT + kt) * 32 + lane) * 16);
        acc = __builtin_amdgcn_wmma_f32_16x16x32_f16(false, a.v, false, b,
                                                     (short)0, acc, false, false);
    }
    return acc;
}

__device__ __forceinline__ void store_gelu_f16(v8f acc, float bias, _Float16* dst,
                                               int ldd, int rowBase, int nt, int lane) {
    const int n  = lane & 15;
    const int mo = (lane & 16) ? 8 : 0;
#pragma unroll
    for (int j = 0; j < 8; ++j) {
        float x = acc[j] + bias;
        dst[(rowBase + mo + j) * ldd + nt * 16 + n] = (_Float16)gelu_erf(x);
    }
}

// ---- weight repack: f32 [Kdim,Ndim] row-major -> f16 WMMA B-fragment order ----
__global__ void pack_w(const float* __restrict__ src, _Float16* __restrict__ dst,
                       int Kdim, int Ndim) {
    int KT = Kdim >> 5, NT = Ndim >> 4;
    int total = KT * NT * 512;   // 32 lanes * 16 halves per tile
    for (int idx = blockIdx.x * blockDim.x + threadIdx.x; idx < total;
         idx += gridDim.x * blockDim.x) {
        int i = idx & 15;
        int l = (idx >> 4) & 31;
        int t = idx >> 9;
        int kt = t % KT, nt = t / KT;
        int k = kt * 32 + ((i & 8) ? 16 : 0) + ((l & 16) ? 8 : 0) + (i & 7);   // B layout: lanes 16-31 hold K=8..15 / K=24..31
        int n = nt * 16 + (l & 15);
        dst[idx] = (_Float16)src[(size_t)k * Ndim + n];
    }
}

// =================== node update ===================
__global__ __launch_bounds__(256)
void node_kernel(const float* __restrict__ hV, const float* __restrict__ hE,
                 const int* __restrict__ Eidx, const float* __restrict__ maskV,
                 const float* __restrict__ maskA,
                 const _Float16* __restrict__ W1p,  const float* __restrict__ W1b,
                 const _Float16* __restrict__ W2p,  const float* __restrict__ W2b,
                 const _Float16* __restrict__ W3p,  const float* __restrict__ W3b,
                 const _Float16* __restrict__ Winp, const float* __restrict__ Winb,
                 const _Float16* __restrict__ Woutp,const float* __restrict__ Woutb,
                 const float* __restrict__ n1g, const float* __restrict__ n1b_,
                 const float* __restrict__ n2g, const float* __restrict__ n2b_,
                 float* __restrict__ hVout) {
    __shared__ __align__(16) unsigned char smem[98432];
    _Float16* sA  = (_Float16*)(smem);            // 64x384 f16 (48KB); aliased by sF 16x512
    _Float16* sM1 = (_Float16*)(smem + 49152);    // 64x128 f16
    _Float16* sM2 = (_Float16*)(smem + 65536);    // 64x128 f16
    float*    sDh = (float*)   (smem + 81920);    // 16x128 f32 (also h2 buffer)
    float*    sHV = (float*)   (smem + 90112);    // 16x128 f32 (post-LN1)
    float*    sSt = (float*)   (smem + 98304);    // mean[16], rstd[16]
    _Float16* sF  = (_Float16*)(smem);            // 16x512 f16 FFN hidden (alias sA)

    const int tid  = threadIdx.x;
    const int lane = tid & 31;
    const int wave = tid >> 5;
    const int nt   = wave;                        // this wave's 16-col tile
    const int n    = lane & 15;
    const int mo   = (lane & 16) ? 8 : 0;
    const int tileBase = blockIdx.x * 16;

    for (int i = tid; i < 16 * HD; i += 256) sDh[i] = 0.0f;
    __syncthreads();

    for (int c = 0; c < 8; ++c) {                 // 8 chunks x 64 message rows
        // build A = [hV(center) | hE | hV(gather)] in f16
        for (int t = tid; t < 64 * 96; t += 256) {
            int r  = t / 96;
            int gq = t - r * 96;                  // float4 group within row
            int lr = c * 64 + r;
            int nl = lr >> 5, kk = lr & 31;
            int g  = tileBase + nl;
            const float* src;
            if (gq < 32)       src = hV + (size_t)g * HD + gq * 4;
            else if (gq < 64)  src = hE + ((size_t)g * KNB + kk) * HD + (gq - 32) * 4;
            else {
                int id = Eidx[g * KNB + kk];
                src = hV + (size_t)id * HD + (gq - 64) * 4;
            }
            float4 v = *(const float4*)src;
            _Float16* d = sA + r * FIN + gq * 4;
            d[0] = (_Float16)v.x; d[1] = (_Float16)v.y;
            d[2] = (_Float16)v.z; d[3] = (_Float16)v.w;
        }
        __syncthreads();
        // layer 1: [64,384]@[384,128] + gelu
        {
            float b = W1b[nt * 16 + n];
            for (int rt = 0; rt < 4; ++rt)
                store_gelu_f16(mma_rowtile(sA, FIN, rt * 16, W1p, KT_FIN, nt, lane),
                               b, sM1, HD, rt * 16, nt, lane);
        }
        __syncthreads();
        // layer 2
        {
            float b = W2b[nt * 16 + n];
            for (int rt = 0; rt < 4; ++rt)
                store_gelu_f16(mma_rowtile(sM1, HD, rt * 16, W2p, KT_H, nt, lane),
                               b, sM2, HD, rt * 16, nt, lane);
        }
        __syncthreads();
        // layer 3 + mask + neighbor reduction
        {
            float b = W3b[nt * 16 + n];
            for (int rt = 0; rt < 4; ++rt) {
                v8f acc = mma_rowtile(sM2, HD, rt * 16, W3p, KT_H, nt, lane);
#pragma unroll
                for (int j = 0; j < 8; ++j) {
                    int gr = c * 64 + rt * 16 + mo + j;
                    int nl = gr >> 5, kk = gr & 31;
                    float ma = maskA[(tileBase + nl) * KNB + kk];
                    atomicAdd(&sDh[nl * HD + nt * 16 + n], (acc[j] + b) * ma);
                }
            }
        }
        __syncthreads();
    }

    // residual + LayerNorm 1
    if (tid < 16) {
        int g = tileBase + tid;
        float s = 0.0f, sq = 0.0f;
        for (int j = 0; j < HD; ++j) {
            float v = hV[(size_t)g * HD + j] + sDh[tid * HD + j] * (1.0f / 30.0f);
            sHV[tid * HD + j] = v;
            s += v; sq += v * v;
        }
        float mean = s * (1.0f / 128.0f);
        float var  = sq * (1.0f / 128.0f) - mean * mean;
        sSt[tid] = mean; sSt[16 + tid] = rsqrtf(var + 1e-5f);
    }
    __syncthreads();
    for (int i = tid; i < 16 * HD; i += 256) {
        int node = i >> 7, j = i & 127;
        float x = (sHV[i] - sSt[node]) * sSt[16 + node] * n1g[j] + n1b_[j];
        sHV[i] = x;
        sM1[i] = (_Float16)x;       // FFN input (16x128)
    }
    __syncthreads();
    // FFN: [16,128]@[128,512] gelu
    for (int q = 0; q < 4; ++q) {
        int ntq = wave + 8 * q;
        store_gelu_f16(mma_rowtile(sM1, HD, 0, Winp, KT_H, ntq, lane),
                       Winb[ntq * 16 + n], sF, 512, 0, ntq, lane);
    }
    __syncthreads();
    // FFN: [16,512]@[512,128] + residual
    {
        v8f acc = mma_rowtile(sF, 512, 0, Woutp, KT_FFN, nt, lane);
        float b = Woutb[nt * 16 + n];
#pragma unroll
        for (int j = 0; j < 8; ++j) {
            int m = mo + j, col = nt * 16 + n;
            sDh[m * HD + col] = acc[j] + b + sHV[m * HD + col];
        }
    }
    __syncthreads();
    // LayerNorm 2 + mask_V + output
    if (tid < 16) {
        float s = 0.0f, sq = 0.0f;
        for (int j = 0; j < HD; ++j) { float v = sDh[tid * HD + j]; s += v; sq += v * v; }
        float mean = s * (1.0f / 128.0f);
        float var  = sq * (1.0f / 128.0f) - mean * mean;
        sSt[tid] = mean; sSt[16 + tid] = rsqrtf(var + 1e-5f);
    }
    __syncthreads();
    for (int i = tid; i < 16 * HD; i += 256) {
        int node = i >> 7, j = i & 127;
        int g = tileBase + node;
        float x = (sDh[i] - sSt[node]) * sSt[16 + node] * n2g[j] + n2b_[j];
        hVout[(size_t)g * HD + j] = maskV[g] * x;
    }
}

// =================== edge update ===================
__global__ __launch_bounds__(256)
void edge_kernel(const float* __restrict__ hVn, const float* __restrict__ hE,
                 const int* __restrict__ Eidx,
                 const _Float16* __restrict__ W11p, const float* __restrict__ W11b,
                 const _Float16* __restrict__ W12p, const float* __restrict__ W12b,
                 const _Float16* __restrict__ W13p, const float* __restrict__ W13b,
                 const float* __restrict__ n3g, const float* __restrict__ n3b_,
                 float* __restrict__ hEout) {
    __shared__ __align__(16) unsigned char smem[86016];
    _Float16* sA  = (_Float16*)(smem);            // 64x384 f16 (48KB)
    _Float16* sM1 = (_Float16*)(smem + 49152);    // 64x128 f16
    _Float16* sM2 = (_Float16*)(smem + 65536);    // 64x128 f16
    float*    sM3 = (float*)   (smem);            // 64x128 f32 (alias sA after use)
    float*    sRs = (float*)   (smem + 81920);    // 64x4 partial sums
    float*    sRq = (float*)   (smem + 82944);    // 64x4 partial sumsq

    const int tid  = threadIdx.x;
    const int lane = tid & 31;
    const int wave = tid >> 5;
    const int nt   = wave;
    const int n    = lane & 15;
    const int mo   = (lane & 16) ? 8 : 0;
    const int tileBase = blockIdx.x * 16;

    for (int c = 0; c < 8; ++c) {
        for (int t = tid; t < 64 * 96; t += 256) {
            int r  = t / 96;
            int gq = t - r * 96;
            int lr = c * 64 + r;
            int nl = lr >> 5, kk = lr & 31;
            int g  = tileBase + nl;
            const float* src;
            if (gq < 32)       src = hVn + (size_t)g * HD + gq * 4;
            else if (gq < 64)  src = hE + ((size_t)g * KNB + kk) * HD + (gq - 32) * 4;
            else {
                int id = Eidx[g * KNB + kk];
                src = hVn + (size_t)id * HD + (gq - 64) * 4;
            }
            float4 v = *(const float4*)src;
            _Float16* d = sA + r * FIN + gq * 4;
            d[0] = (_Float16)v.x; d[1] = (_Float16)v.y;
            d[2] = (_Float16)v.z; d[3] = (_Float16)v.w;
        }
        __syncthreads();
        {
            float b = W11b[nt * 16 + n];
            for (int rt = 0; rt < 4; ++rt)
                store_gelu_f16(mma_rowtile(sA, FIN, rt * 16, W11p, KT_FIN, nt, lane),
                               b, sM1, HD, rt * 16, nt, lane);
        }
        __syncthreads();
        {
            float b = W12b[nt * 16 + n];
            for (int rt = 0; rt < 4; ++rt)
                store_gelu_f16(mma_rowtile(sM1, HD, rt * 16, W12p, KT_H, nt, lane),
                               b, sM2, HD, rt * 16, nt, lane);
        }
        __syncthreads();
        {
            float b = W13b[nt * 16 + n];
            for (int rt = 0; rt < 4; ++rt) {
                v8f acc = mma_rowtile(sM2, HD, rt * 16, W13p, KT_H, nt, lane);
#pragma unroll
                for (int j = 0; j < 8; ++j)
                    sM3[(rt * 16 + mo + j) * HD + nt * 16 + n] = acc[j] + b;
            }
        }
        __syncthreads();
        // per-edge-row LayerNorm of (h_E + m) and write out
        {
            int row  = tid >> 2;     // 0..63
            int part = tid & 3;      // 32 elems each
            int lr = c * 64 + row;
            int nl = lr >> 5, kk = lr & 31;
            size_t eoff = ((size_t)(tileBase + nl) * KNB + kk) * HD;
            float s = 0.0f, sq = 0.0f;
            for (int j = part * 32; j < part * 32 + 32; ++j) {
                float v = hE[eoff + j] + sM3[row * HD + j];
                sM3[row * HD + j] = v;
                s += v; sq += v * v;
            }
            sRs[row * 4 + part] = s;
            sRq[row * 4 + part] = sq;
            __syncthreads();
            float ts = sRs[row * 4] + sRs[row * 4 + 1] + sRs[row * 4 + 2] + sRs[row * 4 + 3];
            float tq = sRq[row * 4] + sRq[row * 4 + 1] + sRq[row * 4 + 2] + sRq[row * 4 + 3];
            float mean = ts * (1.0f / 128.0f);
            float rstd = rsqrtf(tq * (1.0f / 128.0f) - mean * mean + 1e-5f);
            for (int j = part * 32; j < part * 32 + 32; ++j) {
                float v = (sM3[row * HD + j] - mean) * rstd;
                hEout[eoff + j] = v * n3g[j] + n3b_[j];
            }
        }
        __syncthreads();
    }
}

extern "C" void kernel_launch(void* const* d_in, const int* in_sizes, int n_in,
                              void* d_out, int out_size, void* d_ws, size_t ws_size,
                              hipStream_t stream) {
    const float* hV    = (const float*)d_in[0];
    const float* hE    = (const float*)d_in[1];
    const int*   Eidx  = (const int*)  d_in[2];
    const float* maskV = (const float*)d_in[3];
    const float* maskA = (const float*)d_in[4];
    const float* W1w   = (const float*)d_in[5];   const float* W1b  = (const float*)d_in[6];
    const float* W2w   = (const float*)d_in[7];   const float* W2b  = (const float*)d_in[8];
    const float* W3w   = (const float*)d_in[9];   const float* W3b  = (const float*)d_in[10];
    const float* W11w  = (const float*)d_in[11];  const float* W11b = (const float*)d_in[12];
    const float* W12w  = (const float*)d_in[13];  const float* W12b = (const float*)d_in[14];
    const float* W13w  = (const float*)d_in[15];  const float* W13b = (const float*)d_in[16];
    const float* Winw  = (const float*)d_in[17];  const float* Winb = (const float*)d_in[18];
    const float* Woutw = (const float*)d_in[19];  const float* Woutb= (const float*)d_in[20];
    const float* n1g   = (const float*)d_in[21];  const float* n1b  = (const float*)d_in[22];
    const float* n2g   = (const float*)d_in[23];  const float* n2b  = (const float*)d_in[24];
    const float* n3g   = (const float*)d_in[25];  const float* n3b  = (const float*)d_in[26];

    _Float16* ws = (_Float16*)d_ws;
    _Float16* W1p   = ws + OFF_W1;
    _Float16* W2p   = ws + OFF_W2;
    _Float16* W3p   = ws + OFF_W3;
    _Float16* W11p  = ws + OFF_W11;
    _Float16* W12p  = ws + OFF_W12;
    _Float16* W13p  = ws + OFF_W13;
    _Float16* Winp  = ws + OFF_WIN;
    _Float16* Woutp = ws + OFF_WOUT;

    pack_w<<<96, 256, 0, stream>>>(W1w,   W1p,   384, 128);
    pack_w<<<64, 256, 0, stream>>>(W2w,   W2p,   128, 128);
    pack_w<<<64, 256, 0, stream>>>(W3w,   W3p,   128, 128);
    pack_w<<<96, 256, 0, stream>>>(W11w,  W11p,  384, 128);
    pack_w<<<64, 256, 0, stream>>>(W12w,  W12p,  128, 128);
    pack_w<<<64, 256, 0, stream>>>(W13w,  W13p,  128, 128);
    pack_w<<<128, 256, 0, stream>>>(Winw, Winp,  128, 512);
    pack_w<<<128, 256, 0, stream>>>(Woutw, Woutp, 512, 128);

    float* hVout = (float*)d_out;
    float* hEout = (float*)d_out + (size_t)NNODE * HD;

    node_kernel<<<NNODE / 16, 256, 0, stream>>>(hV, hE, Eidx, maskV, maskA,
        W1p, W1b, W2p, W2b, W3p, W3b, Winp, Winb, Woutp, Woutb,
        n1g, n1b, n2g, n2b, hVout);

    edge_kernel<<<NNODE / 16, 256, 0, stream>>>(hVout, hE, Eidx,
        W11p, W11b, W12p, W12b, W13p, W13b, n3g, n3b, hEout);
}
